// PointnetFPModule_72052371357928
// MI455X (gfx1250) — compile-verified, matrix operand order checked by reading, not run on hardware
//
#include <hip/hip_runtime.h>
#include <hip/hip_bf16.h>

typedef __attribute__((ext_vector_type(16))) _Float16 v16h;
typedef __attribute__((ext_vector_type(8)))  _Float16 h8;
typedef __attribute__((ext_vector_type(8)))  float    v8f;

#define BN_EPS 1e-5f
#define BATCH 16
#define NQ    4096
#define MK    1024
#define C1    128
#define C2    256
#define CIN   384   // C1 + C2
#define H1    256
#define H2    128

// ---------------------------------------------------------------------------
// Load a 16x32 f16 WMMA operand fragment (A, or B given as W^T rows).
// Lane L holds matrix row (L & 15); K chunks: [K0, K0+8) and [K0+16, K0+24)
// with K0 = (L >> 4) * 8, per the CDNA5 16-bit A-matrix VGPR layout.
// Two 16-byte contiguous loads -> global_load_b128 pairs.
// ---------------------------------------------------------------------------
__device__ __forceinline__ v16h load_frag(const _Float16* __restrict__ p) {
    h8 c0 = *reinterpret_cast<const h8*>(p);
    h8 c1 = *reinterpret_cast<const h8*>(p + 16);
    v16h a;
#pragma unroll
    for (int i = 0; i < 8; ++i) { a[i] = c0[i]; a[i + 8] = c1[i]; }
    return a;
}

// ---------------------------------------------------------------------------
// prep: fold BN into per-channel alpha/beta; cast weights to f16.
//   y = relu((acc + b - mu) * g*rsqrt(v+eps) + bt) = relu(acc*alpha + beta)
// ---------------------------------------------------------------------------
__global__ void prep_kernel(const float* __restrict__ W1, const float* __restrict__ b1,
                            const float* __restrict__ g1, const float* __restrict__ bt1,
                            const float* __restrict__ mu1, const float* __restrict__ v1,
                            const float* __restrict__ W2, const float* __restrict__ b2,
                            const float* __restrict__ g2, const float* __restrict__ bt2,
                            const float* __restrict__ mu2, const float* __restrict__ v2,
                            _Float16* __restrict__ W1h, _Float16* __restrict__ W2h,
                            float* __restrict__ a1, float* __restrict__ be1,
                            float* __restrict__ a2, float* __restrict__ be2) {
    int i = blockIdx.x * blockDim.x + threadIdx.x;
    if (i < H1 * CIN) W1h[i] = (_Float16)W1[i];
    if (i < H2 * H1)  W2h[i] = (_Float16)W2[i];
    if (i < H1) {
        float s = g1[i] * rsqrtf(v1[i] + BN_EPS);
        a1[i] = s; be1[i] = (b1[i] - mu1[i]) * s + bt1[i];
    }
    if (i < H2) {
        float s = g2[i] * rsqrtf(v2[i] + BN_EPS);
        a2[i] = s; be2[i] = (b2[i] - mu2[i]) * s + bt2[i];
    }
}

// ---------------------------------------------------------------------------
// three_nn + three_interpolate + concat -> x (f16, row-major (B*n, 384)).
// One thread per query point; known coords staged in LDS (12 KB); all lanes
// read the same known point per iteration -> LDS broadcast.
// ---------------------------------------------------------------------------
__global__ __launch_bounds__(256)
void three_nn_interp_kernel(const float* __restrict__ unknown,
                            const float* __restrict__ known,
                            const float* __restrict__ unknow_feats,
                            const float* __restrict__ known_feats,
                            _Float16* __restrict__ xh) {
    __shared__ float skx[MK], sky[MK], skz[MK];
    const int b   = blockIdx.y;
    const int tid = threadIdx.x;
    const float* kb = known + (size_t)b * MK * 3;
    for (int i = tid; i < MK; i += 256) {
        skx[i] = kb[i * 3 + 0];
        sky[i] = kb[i * 3 + 1];
        skz[i] = kb[i * 3 + 2];
    }
    __syncthreads();

    const int    q   = blockIdx.x * 256 + tid;
    const size_t row = (size_t)b * NQ + q;
    const float ux = unknown[row * 3 + 0];
    const float uy = unknown[row * 3 + 1];
    const float uz = unknown[row * 3 + 2];

    float d0 = 1e30f, d1 = 1e30f, d2 = 1e30f;
    int   i0 = 0,     i1 = 0,     i2 = 0;
    for (int j = 0; j < MK; ++j) {
        float dx = ux - skx[j], dy = uy - sky[j], dz = uz - skz[j];
        float d  = dx * dx + dy * dy + dz * dz;
        if (d < d0)      { d2 = d1; i2 = i1; d1 = d0; i1 = i0; d0 = d; i0 = j; }
        else if (d < d1) { d2 = d1; i2 = i1; d1 = d;  i1 = j; }
        else if (d < d2) { d2 = d;  i2 = j; }
    }
    float r0 = 1.0f / (d0 + 1e-10f);
    float r1 = 1.0f / (d1 + 1e-10f);
    float r2 = 1.0f / (d2 + 1e-10f);
    float rs = 1.0f / (r0 + r1 + r2);
    float w0 = r0 * rs, w1 = r1 * rs, w2 = r2 * rs;

    const float* f0 = known_feats + ((size_t)b * MK + i0) * C2;
    const float* f1 = known_feats + ((size_t)b * MK + i1) * C2;
    const float* f2 = known_feats + ((size_t)b * MK + i2) * C2;
    _Float16* xr = xh + row * CIN;
#pragma unroll 4
    for (int c = 0; c < C2; ++c)
        xr[c] = (_Float16)(w0 * f0[c] + w1 * f1[c] + w2 * f2[c]);
    const float* uf = unknow_feats + row * C1;
#pragma unroll 4
    for (int c = 0; c < C1; ++c)
        xr[C2 + c] = (_Float16)uf[c];
}

// ---------------------------------------------------------------------------
// GEMM: out[m][o] = relu(alpha[o] * sum_k X[m][k]*W[o][k] + beta[o])
// X: (M, K) f16 row-major, W: (N, K) f16 row-major (== B^T, columns contiguous)
// Block = 256 threads = 8 wave32; block tile = 128(M) x 64(N);
// each wave: 16(M) x 64(N) via 4 accumulators, K stepped by 32 with
// v_wmma_f32_16x16x32_f16. M%128==0, N%64==0, K%32==0 guaranteed.
// ---------------------------------------------------------------------------
__global__ __launch_bounds__(256)
void gemm_bn_relu_kernel(const _Float16* __restrict__ X, const _Float16* __restrict__ W,
                         const float* __restrict__ alpha, const float* __restrict__ beta,
                         int N, int K,
                         _Float16* __restrict__ outh, float* __restrict__ outf) {
    const int lane  = threadIdx.x & 31;
    const int wave  = threadIdx.x >> 5;
    const int row16 = lane & 15;          // A row / B column / D column
    const int kbase = (lane >> 4) << 3;   // 0 or 8
    const long m0 = (long)blockIdx.y * 128 + wave * 16;
    const int  n0 = blockIdx.x * 64;

    const _Float16* xrow = X + (m0 + row16) * (long)K + kbase;
    const _Float16* w0p  = W + (long)(n0 +  0 + row16) * K + kbase;
    const _Float16* w1p  = W + (long)(n0 + 16 + row16) * K + kbase;
    const _Float16* w2p  = W + (long)(n0 + 32 + row16) * K + kbase;
    const _Float16* w3p  = W + (long)(n0 + 48 + row16) * K + kbase;

    v8f acc[4] = {v8f{}, v8f{}, v8f{}, v8f{}};
    for (int k = 0; k < K; k += 32) {
        __builtin_prefetch(xrow + k + 64, 0, 0);   // global_prefetch_b8 on A stream
        v16h a  = load_frag(xrow + k);
        v16h b0 = load_frag(w0p + k);
        v16h b1 = load_frag(w1p + k);
        v16h b2 = load_frag(w2p + k);
        v16h b3 = load_frag(w3p + k);
        acc[0] = __builtin_amdgcn_wmma_f32_16x16x32_f16(false, a, false, b0, (short)0, acc[0], false, false);
        acc[1] = __builtin_amdgcn_wmma_f32_16x16x32_f16(false, a, false, b1, (short)0, acc[1], false, false);
        acc[2] = __builtin_amdgcn_wmma_f32_16x16x32_f16(false, a, false, b2, (short)0, acc[2], false, false);
        acc[3] = __builtin_amdgcn_wmma_f32_16x16x32_f16(false, a, false, b3, (short)0, acc[3], false, false);
    }

    // D layout: VGPR r of lane L -> M = r + 8*(L>>4), N = L&15
    const int mb = (lane >> 4) << 3;
#pragma unroll
    for (int j = 0; j < 4; ++j) {
        const int col = n0 + j * 16 + row16;
        const float al = alpha[col];
        const float be = beta[col];
#pragma unroll
        for (int r = 0; r < 8; ++r) {
            const long m = m0 + mb + r;
            float v = fmaxf(acc[j][r] * al + be, 0.0f);
            if (outh) outh[m * (long)N + col] = (_Float16)v;
            else      outf[m * (long)N + col] = v;
        }
    }
}

// ---------------------------------------------------------------------------
extern "C" void kernel_launch(void* const* d_in, const int* in_sizes, int n_in,
                              void* d_out, int out_size, void* d_ws, size_t ws_size,
                              hipStream_t stream) {
    const float* unknown      = (const float*)d_in[0];
    const float* known        = (const float*)d_in[1];
    const float* unknow_feats = (const float*)d_in[2];
    const float* known_feats  = (const float*)d_in[3];
    const float* W1  = (const float*)d_in[4];
    const float* b1  = (const float*)d_in[5];
    const float* g1  = (const float*)d_in[6];
    const float* bt1 = (const float*)d_in[7];
    const float* mu1 = (const float*)d_in[8];
    const float* v1  = (const float*)d_in[9];
    const float* W2  = (const float*)d_in[10];
    const float* b2  = (const float*)d_in[11];
    const float* g2  = (const float*)d_in[12];
    const float* bt2 = (const float*)d_in[13];
    const float* mu2 = (const float*)d_in[14];
    const float* v2  = (const float*)d_in[15];

    char* ws = (char*)d_ws;
    size_t off = 0;
    auto carve = [&](size_t bytes) -> char* {
        char* p = ws + off;
        off = (off + bytes + 255) & ~(size_t)255;
        return p;
    };
    const long M = (long)BATCH * NQ;                           // 65536
    _Float16* xh   = (_Float16*)carve((size_t)M * CIN * 2);    // 50.3 MB
    _Float16* y1h  = (_Float16*)carve((size_t)M * H1 * 2);     // 33.5 MB
    _Float16* W1h  = (_Float16*)carve((size_t)H1 * CIN * 2);
    _Float16* W2h  = (_Float16*)carve((size_t)H2 * H1 * 2);
    float*    a1v  = (float*)carve(H1 * 4);
    float*    be1v = (float*)carve(H1 * 4);
    float*    a2v  = (float*)carve(H2 * 4);
    float*    be2v = (float*)carve(H2 * 4);

    // 1) fold BN, cast weights
    prep_kernel<<<dim3((H1 * CIN + 255) / 256), dim3(256), 0, stream>>>(
        W1, b1, g1, bt1, mu1, v1, W2, b2, g2, bt2, mu2, v2,
        W1h, W2h, a1v, be1v, a2v, be2v);

    // 2) three_nn + interpolate + concat -> xh (f16, (B*n, 384))
    three_nn_interp_kernel<<<dim3(NQ / 256, BATCH), dim3(256), 0, stream>>>(
        unknown, known, unknow_feats, known_feats, xh);

    // 3) layer 1: (65536,384) x (384,256) + BN + ReLU -> y1h (f16)
    gemm_bn_relu_kernel<<<dim3(H1 / 64, M / 128), dim3(256), 0, stream>>>(
        xh, W1h, a1v, be1v, H1, CIN, y1h, nullptr);

    // 4) layer 2: (65536,256) x (256,128) + BN + ReLU -> d_out (f32)
    gemm_bn_relu_kernel<<<dim3(H2 / 64, M / 128), dim3(256), 0, stream>>>(
        y1h, W2h, a2v, be2v, H2, H1, nullptr, (float*)d_out);
}
